// SandayASR_45105746542663
// MI455X (gfx1250) — matrix operand order repaired and checked by reading, not compiled, not run on hardware
//
#include <hip/hip_runtime.h>
#include <cmath>
#include <cstdint>

// ---------------------------------------------------------------------------
// DREAM recurrent stack on MI455X (gfx1250):
//   - persistent kernel, 3 workgroups (one per layer), weights in LDS (f16)
//   - all matmuls via v_wmma_f32_16x16x32_f16 (f32 accumulate)
//   - fragment-major LDS layouts: every WMMA operand is ONE aligned v16h load
//     at lane*32B -> conflict-free banks, immediate-offset ds_load_b128
//   - cross-layer handoff via GLOBAL_LOAD_ASYNC_TO_LDS_B128 (ASYNCcnt)
//   - CTC projection + log_softmax as a second, parallel kernel
// ---------------------------------------------------------------------------

typedef _Float16 h16;
typedef __attribute__((ext_vector_type(16))) _Float16 v16h;
typedef __attribute__((ext_vector_type(8)))  _Float16 v8h;
typedef __attribute__((ext_vector_type(8)))  float    v8f;

#define NB   32
#define NT   2000
#define DIN  80
#define NH   256
#define NR   16
#define NP   45
#define KPAD 512      // [x(256, zero-padded) | h(256)]
#define KFRG (KPAD / 32)   // 16 k-fragments per tile row (panel / Wt)
#define LAMW 0.9f
#define ETAW 0.5f

struct RecParams {
  const float* x;
  const float* win[3];
  const float* wrec[3];
  const float* bias[3];
  const float* u[3];
  const float* v[3];
  const float* tau[3];
  h16* stage;   // [2 producer layers][2 buffers][NB*NH] f16
  h16* hid;     // [NT][NB][NH] f16 (layer-2 output)
  int* flags;   // [0..2]=progress, [4..6]=consumer acks
};

static __device__ __forceinline__ v16h cat8(v8h lo, v8h hi) {
  return __builtin_shufflevector(lo, hi, 0,1,2,3,4,5,6,7,8,9,10,11,12,13,14,15);
}

// ---- fragment-major scatter addresses (in halves) -------------------------
// One fragment = 32 lanes x 16 halves = 512 halves; fragment f at f*512;
// lane L's 16 halves contiguous at L*16.

// A-operand scatter (16x32 f16 per wave): lane L holds row m=L&15,
// halves 0..7 = K kk+khb+0..7, halves 8..15 = K kk+16+khb+0..7, khb=(L<16?0:8)
static __device__ __forceinline__ int aIdx(int row, int k) {
  int f   = (row >> 4) * KFRG + (k >> 5);
  int sub = k & 31;
  int ln  = (row & 15) + (((sub >> 3) & 1) << 4);
  int e   = ((sub >> 4) << 3) | (sub & 7);
  return f * 512 + ln * 16 + e;
}
// B-operand scatter (32x16 f16 per wave): lane L holds col n=L&15,
// halves 0..7 = K kk+khb+0..7, halves 8..15 = K kk+16+khb+0..7, khb=(L<16?0:8)
static __device__ __forceinline__ int bIdx(int n, int k, int kfrags) {
  int f   = (n >> 4) * kfrags + (k >> 5);
  int sub = k & 31;
  int ln  = (n & 15) + (((sub >> 3) & 1) << 4);
  int e   = ((sub >> 4) << 3) | (sub & 7);
  return f * 512 + ln * 16 + e;
}

__global__ __launch_bounds__(256, 1) void dream_rec(RecParams p) {
  // ---- LDS (~322 KB, fits CDNA5's 320KB-per-WGP budget) ----
  __shared__ __align__(32) h16 sWt[NH * KPAD];   // 256 KB: [win;wrec]^T, fragment-major
  __shared__ __align__(32) h16 sXH[NB * KPAD];   //  32 KB: [x|h] panel, fragment-major
  __shared__ __align__(32) h16 sUt[NR * NH];     //   8 KB: u as B-fragments
  __shared__ __align__(32) h16 sVt[NR * NH];     //   8 KB: v as B-fragments
  __shared__ __align__(32) h16 sVrm[NH * NR];    //   8 KB: v row-major [h][r]
  __shared__ __align__(32) h16 sK[NB * NR];
  __shared__ __align__(32) h16 sQ[NB * NR];
  __shared__ __align__(32) h16 sM16[NB * NR];    // m = k . M (f16)
  __shared__ __align__(32) h16 sZero[16];

  const int layer = blockIdx.x;
  const int tid   = threadIdx.x;
  const int wave  = tid >> 5;
  const int lane  = tid & 31;
  const bool lo16 = (lane < 16);
  const int rhalf = lo16 ? 0 : 8;   // A-fragment K sub-offset (halves)
  const int hoff  = lo16 ? 0 : 8;   // C/D row sub-offset
  const int ncol  = lane & 15;

  const float* win  = p.win[layer];
  const float* wrec = p.wrec[layer];
  const float* bias = p.bias[layer];
  const float* uW   = p.u[layer];
  const float* vW   = p.v[layer];
  const float* tau  = p.tau[layer];
  const int din = (layer == 0) ? DIN : NH;

  // ---- one-time LDS fill: f32 -> f16, fragment-major scatter ----
  for (int i = tid; i < NH * KPAD; i += 256) {
    int n = i >> 9, k = i & (KPAD - 1);
    float w;
    if (k < NH) w = (k < din) ? win[k * NH + n] : 0.f;   // x-part (zero-padded)
    else        w = wrec[(k - NH) * NH + n];             // h-part
    sWt[bIdx(n, k, KFRG)] = (h16)w;
  }
  for (int i = tid; i < NR * NH; i += 256) {
    int r = i >> 8, hh = i & 255;
    sUt[bIdx(r, hh, NH / 32)] = (h16)uW[hh * NR + r];
    sVt[bIdx(r, hh, NH / 32)] = (h16)vW[hh * NR + r];
    sVrm[i] = (h16)vW[i];            // v is (H,R) row-major already
  }
  for (int i = tid; i < NB * KPAD; i += 256) sXH[i] = (h16)0.f;  // zero: layout-invariant
  if (tid < 16) sZero[tid] = (h16)0.f;

  // ---- per-thread tile assignment: wave -> (mt, 4 n-tiles) ----
  const int mt = wave & 1;
  const int ntBase = (wave >> 1) * 4;

  float hreg[4][8];                  // h state, matches C/D fragment layout
  float invtau[4], br[4];
  #pragma unroll
  for (int j = 0; j < 4; ++j) {
    int n = (ntBase + j) * 16 + ncol;
    invtau[j] = 1.f / tau[n];
    br[j] = bias[n];
    #pragma unroll
    for (int i = 0; i < 8; ++i) hreg[j][i] = 0.f;
  }

  // ---- fast-weight state: thread owns M[b][:, sc0:sc0+2] (column pair) ----
  const int mb  = tid >> 3;
  const int sc0 = (tid & 7) * 2;
  float Mreg[NR][2];
  #pragma unroll
  for (int r = 0; r < NR; ++r) { Mreg[r][0] = 0.f; Mreg[r][1] = 0.f; }

  __syncthreads();

  for (int t = 0; t < NT; ++t) {
    // ---------------- Phase A: acquire input (layer 0: x-region; layers 1-2: h-region) ------------------
    if (layer == 0) {
      for (int i = tid; i < NB * DIN; i += 256) {
        int b = i / DIN, d = i - b * DIN;
        sXH[aIdx(b, d)] = (h16)p.x[((size_t)b * NT + t) * DIN + d];
      }
      if (t + 1 < NT && tid < NB)
        __builtin_prefetch(&p.x[((size_t)tid * NT + (t + 1)) * DIN], 0, 3);
      __syncthreads();
    } else {
      if (tid == 0) {
        while (__hip_atomic_load(&p.flags[layer - 1], __ATOMIC_ACQUIRE,
                                 __HIP_MEMORY_SCOPE_AGENT) < t + 1)
          __builtin_amdgcn_s_sleep(1);
      }
      __syncthreads();
      // CDNA5 async DMA: global -> LDS, one 16B chunk per op (ASYNCcnt-tracked).
      // Every aligned 8-half k-group is contiguous in the A-fragment scatter.
      const h16* st = p.stage + (size_t)(((layer - 1) * 2) + (t & 1)) * (NB * NH);
      for (int i = tid * 8; i < NB * NH; i += 256 * 8) {
        int b = i >> 8, kh = i & 255;
        unsigned ldsoff = (unsigned)(uintptr_t)&sXH[aIdx(b, NH + kh)];
        uint64_t gaddr  = (uint64_t)(uintptr_t)(st + i);
        asm volatile("global_load_async_to_lds_b128 %0, %1, off"
                     :: "v"(ldsoff), "v"(gaddr) : "memory");
      }
      asm volatile("s_wait_asynccnt 0" ::: "memory");
      __syncthreads();
      if (tid == 0)
        __hip_atomic_store(&p.flags[4 + layer], t + 1, __ATOMIC_RELEASE,
                           __HIP_MEMORY_SCOPE_AGENT);   // ack: buffer consumed
    }

    // ---------------- Phase B: k = h@u, q = h@v (WMMA, waves 0..3) ----------
    if (wave < 4) {
      const h16* Bm = (wave < 2) ? sUt : sVt;
      const int mtk = wave & 1;
      const h16* aBase = &sXH[(mtk * KFRG + 8) * 512 + lane * 16]; // h-region frags
      const h16* bBase = &Bm[lane * 16];
      v8f acc = {};
      #pragma unroll
      for (int kk = 0; kk < NH; kk += 32) {
        v16h a  = *(const v16h*)(aBase + (kk >> 5) * 512);
        v16h bf = *(const v16h*)(bBase + (kk >> 5) * 512);
        acc = __builtin_amdgcn_wmma_f32_16x16x32_f16(false, a, false, bf,
                                                     (short)0, acc, false, false);
      }
      h16* dst = (wave < 2) ? sK : sQ;
      #pragma unroll
      for (int i = 0; i < 8; ++i)
        dst[(mtk * 16 + i + hoff) * NR + ncol] = (h16)acc[i];
    }
    __syncthreads();

    // ------- Phase C: m[b,s] = sum_r k*M ; M = LAM*M + ETA*k (x) q ----------
    {
      float kv[NR];
      #pragma unroll
      for (int r = 0; r < NR; ++r) kv[r] = (float)sK[mb * NR + r];
      float q0 = (float)sQ[mb * NR + sc0];
      float q1 = (float)sQ[mb * NR + sc0 + 1];
      float m0 = 0.f, m1 = 0.f;
      #pragma unroll
      for (int r = 0; r < NR; ++r) {
        m0 += kv[r] * Mreg[r][0];
        m1 += kv[r] * Mreg[r][1];
        Mreg[r][0] = LAMW * Mreg[r][0] + ETAW * kv[r] * q0;
        Mreg[r][1] = LAMW * Mreg[r][1] + ETAW * kv[r] * q1;
      }
      sM16[mb * NR + sc0]     = (h16)m0;
      sM16[mb * NR + sc0 + 1] = (h16)m1;
    }
    __syncthreads();

    // ------- Phase D: acc = m @ v^T  (K=16 padded to 32) ; acc += panel@Wt --
    v8f acc[4] = {};
    {
      const int arow = (mt * 16 + ncol) * NR;
      v8h z8 = {};
      v16h am = cat8(*(const v8h*)&sM16[arow + rhalf], z8);  // K>=16 is zero
      #pragma unroll
      for (int j = 0; j < 4; ++j) {
        int n = (ntBase + j) * 16 + ncol;
        const h16* bp = &sVrm[n * NR + rhalf];               // K>=16 rows = 0
        v16h bf = cat8(*(const v8h*)bp, z8);
        acc[j] = __builtin_amdgcn_wmma_f32_16x16x32_f16(false, am, false, bf,
                                                        (short)0, acc[j], false, false);
      }
    }
    {
      const h16* aBase = &sXH[(mt * KFRG) * 512 + lane * 16];
      const h16* bBase0 = &sWt[((ntBase + 0) * KFRG) * 512 + lane * 16];
      const h16* bBase1 = &sWt[((ntBase + 1) * KFRG) * 512 + lane * 16];
      const h16* bBase2 = &sWt[((ntBase + 2) * KFRG) * 512 + lane * 16];
      const h16* bBase3 = &sWt[((ntBase + 3) * KFRG) * 512 + lane * 16];
      #pragma unroll
      for (int kk = 0; kk < KPAD; kk += 32) {
        const int fo = (kk >> 5) * 512;
        v16h a = *(const v16h*)(aBase + fo);
        acc[0] = __builtin_amdgcn_wmma_f32_16x16x32_f16(false, a, false,
                   *(const v16h*)(bBase0 + fo), (short)0, acc[0], false, false);
        acc[1] = __builtin_amdgcn_wmma_f32_16x16x32_f16(false, a, false,
                   *(const v16h*)(bBase1 + fo), (short)0, acc[1], false, false);
        acc[2] = __builtin_amdgcn_wmma_f32_16x16x32_f16(false, a, false,
                   *(const v16h*)(bBase2 + fo), (short)0, acc[2], false, false);
        acc[3] = __builtin_amdgcn_wmma_f32_16x16x32_f16(false, a, false,
                   *(const v16h*)(bBase3 + fo), (short)0, acc[3], false, false);
      }
    }
    __syncthreads();

    // ------- back-pressure: don't overwrite stage buffer consumer still reads
    if (layer < 2 && t >= 2) {
      if (tid == 0) {
        while (__hip_atomic_load(&p.flags[4 + layer + 1], __ATOMIC_ACQUIRE,
                                 __HIP_MEMORY_SCOPE_AGENT) < t - 1)
          __builtin_amdgcn_s_sleep(1);
      }
    }
    __syncthreads();

    // ------- Phase E: h <- h + (tanh(pre) - h)/tau ; publish ---------------
    h16* stageOut = (layer < 2)
        ? (p.stage + (size_t)((layer * 2) + (t & 1)) * (NB * NH)) : nullptr;
    h16* hidOut = p.hid + (size_t)t * (NB * NH);
    #pragma unroll
    for (int j = 0; j < 4; ++j) {
      int n = (ntBase + j) * 16 + ncol;
      #pragma unroll
      for (int i = 0; i < 8; ++i) {
        int b = mt * 16 + i + hoff;
        float pre = acc[j][i] + br[j];
        float hc = tanhf(pre);
        float hn = hreg[j][i] + (hc - hreg[j][i]) * invtau[j];
        hreg[j][i] = hn;
        h16 hv = (h16)hn;
        sXH[aIdx(b, NH + n)] = hv;            // next step's A operand
        if (layer < 2) stageOut[b * NH + n] = hv;
        else           hidOut[b * NH + n]  = hv;
      }
    }
    __syncthreads();
    if (tid == 0) {
      __threadfence();
      __hip_atomic_store(&p.flags[layer], t + 1, __ATOMIC_RELEASE,
                         __HIP_MEMORY_SCOPE_AGENT);
    }
  }
}

// ---------------------------------------------------------------------------
// CTC head: one wave per (b,t) row; lane covers classes {lane, lane+32};
// wave-shuffle log-softmax reduction.
// ---------------------------------------------------------------------------
__global__ __launch_bounds__(256) void ctc_head(const h16* __restrict__ hid,
                                                const float* __restrict__ w,
                                                const float* __restrict__ bv,
                                                float* __restrict__ logits,
                                                float* __restrict__ logp) {
  const int wave = threadIdx.x >> 5, lane = threadIdx.x & 31;
  const long row = (long)blockIdx.x * 8 + wave;
  if (row >= (long)NB * NT) return;
  const int b = (int)(row / NT);
  const int t = (int)(row - (long)b * NT);
  const h16* hrow = hid + ((size_t)t * NB + b) * NH;
  const int p0 = lane, p1 = lane + 32;
  const bool p1v = (p1 < NP);
  float l0 = bv[p0];
  float l1 = p1v ? bv[p1] : 0.f;
  for (int k = 0; k < NH; ++k) {
    float hv = (float)hrow[k];
    l0 += hv * w[k * NP + p0];
    l1 += hv * w[k * NP + (p1v ? p1 : 0)];
  }
  float vmax = fmaxf(l0, p1v ? l1 : -__builtin_inff());
  #pragma unroll
  for (int off = 16; off > 0; off >>= 1)
    vmax = fmaxf(vmax, __shfl_xor(vmax, off, 32));
  float s = expf(l0 - vmax) + (p1v ? expf(l1 - vmax) : 0.f);
  #pragma unroll
  for (int off = 16; off > 0; off >>= 1)
    s += __shfl_xor(s, off, 32);
  float logZ = vmax + logf(s);
  size_t o = (size_t)row * NP;
  logits[o + p0] = l0;
  logp[o + p0]   = l0 - logZ;
  if (p1v) {
    logits[o + p1] = l1;
    logp[o + p1]   = l1 - logZ;
  }
}

extern "C" void kernel_launch(void* const* d_in, const int* in_sizes, int n_in,
                              void* d_out, int out_size, void* d_ws, size_t ws_size,
                              hipStream_t stream) {
  (void)in_sizes; (void)n_in; (void)out_size; (void)ws_size;

  RecParams rp;
  rp.x = (const float*)d_in[0];
  for (int l = 0; l < 3; ++l) {
    rp.win[l]  = (const float*)d_in[1 + 6 * l];
    rp.wrec[l] = (const float*)d_in[2 + 6 * l];
    rp.bias[l] = (const float*)d_in[3 + 6 * l];
    rp.u[l]    = (const float*)d_in[4 + 6 * l];
    rp.v[l]    = (const float*)d_in[5 + 6 * l];
    rp.tau[l]  = (const float*)d_in[6 + 6 * l];
  }
  const float* ctc_w = (const float*)d_in[19];
  const float* ctc_b = (const float*)d_in[20];

  // workspace: [0,256) flags | [256, 256+64K) stage | then hid (f16, 32.8 MB)
  char* ws = (char*)d_ws;
  rp.flags = (int*)ws;
  rp.stage = (h16*)(ws + 256);
  rp.hid   = (h16*)(ws + 256 + (size_t)4 * NB * NH * sizeof(h16));

  hipMemsetAsync(d_ws, 0, 256, stream);                 // reset pipeline flags
  dream_rec<<<3, 256, 0, stream>>>(rp);                 // 3 WGPs, persistent
  float* logits = (float*)d_out;
  float* logp   = logits + (size_t)NB * NT * NP;
  ctc_head<<<(NB * NT) / 8, 256, 0, stream>>>(rp.hid, ctc_w, ctc_b, logits, logp);
}
